// ImageNTXentLoss_61280593379703
// MI455X (gfx1250) — matrix operand order repaired and checked by reading, not compiled, not run on hardware
//
#include <hip/hip_runtime.h>
#include <hip/hip_bf16.h>
#include <math.h>

typedef __attribute__((ext_vector_type(16))) _Float16 v16h;
typedef __attribute__((ext_vector_type(8)))  _Float16 v8h;
typedef __attribute__((ext_vector_type(8)))  float    v8f;

#define N_ROWS   4096
#define N2       8192
#define DDIM     768
#define INV_T    (1.0f / 0.07f)
#define NEG_BIG  (-3.0e38f)

#define ROWS_PER_BLK 64        // 4 waves x 16 rows
#define COLS_PER_TILE 128      // LDS-staged column tile
#define NSPLIT 4               // column splits (online-softmax partials)
#define COLS_PER_SPLIT (N2 / NSPLIT)          // 2048
#define NCB (COLS_PER_SPLIT / COLS_PER_TILE)  // 16 column tiles per split
#define KSTEP 32               // K per WMMA (f16)
#define KCHUNK 128             // halves per LDS chunk = 4 WMMA k-steps
#define NKCH (DDIM / KCHUNK)   // 6 chunks over K
#define NCHUNK (NCB * NKCH)    // 96 chunks total per split
#define LDS_STRIDE 136         // padded halves per column (272 B, 16B aligned)

union FragU { v16h v; v8h h[2]; };

// ---------------------------------------------------------------------------
// Kernel 1: L2-normalize each row of [z1; z2] and store as f16
// ---------------------------------------------------------------------------
__global__ __launch_bounds__(256) void ntxent_norm_kernel(
    const float* __restrict__ z1, const float* __restrict__ z2,
    _Float16* __restrict__ zh) {
  const int row = blockIdx.x;
  const float* src = (row < N_ROWS) ? (z1 + (size_t)row * DDIM)
                                    : (z2 + (size_t)(row - N_ROWS) * DDIM);
  __shared__ float red[256];
  float p = 0.f;
  for (int j = threadIdx.x; j < DDIM; j += 256) { float x = src[j]; p += x * x; }
  red[threadIdx.x] = p;
  __syncthreads();
  for (int s = 128; s > 0; s >>= 1) {
    if (threadIdx.x < s) red[threadIdx.x] += red[threadIdx.x + s];
    __syncthreads();
  }
  const float nrm = sqrtf(red[0]);
  const float scale = 1.0f / fmaxf(nrm, 1e-12f);
  _Float16* dst = zh + (size_t)row * DDIM;
  for (int j = threadIdx.x; j < DDIM; j += 256)
    dst[j] = (_Float16)(src[j] * scale);
}

// ---------------------------------------------------------------------------
// Kernel 2: fused WMMA GEMM (sim = Z Z^T / T) + online masked softmax partials
// grid.x = row blocks (64 rows), grid.y = column split (2048 cols each)
// 128 threads = 4 waves; wave w owns rows [blk*64 + w*16, +16)
// B tiles stream global->LDS via async DMA (ASYNCcnt), double buffered.
// ---------------------------------------------------------------------------
__global__ __launch_bounds__(128) void ntxent_sim_kernel(
    const _Float16* __restrict__ zh,
    float* __restrict__ pM, float* __restrict__ pS, float* __restrict__ pPos) {
  __shared__ _Float16 lb[2][COLS_PER_TILE * LDS_STRIDE];   // 2 x 34 KB

  const int tid  = threadIdx.x;
  const int wave = tid >> 5;
  const int lane = tid & 31;
  const int l16  = lane & 15;
  const int hi   = lane >> 4;              // 0: lanes 0-15, 1: lanes 16-31

  const int rowBase   = blockIdx.x * ROWS_PER_BLK + wave * 16;
  const int splitBase = blockIdx.y * COLS_PER_SPLIT;
  const int myRow     = rowBase + l16;     // A-matrix row for this lane
  const _Float16* aBase = zh + (size_t)myRow * DDIM + hi * 8;

  float runM[8], runS[8], posA[8];
#pragma unroll
  for (int i = 0; i < 8; ++i) { runM[i] = NEG_BIG; runS[i] = 0.f; posA[i] = 0.f; }

  // Async global->LDS fill of chunk g into lb[g&1]: 128 cols x 128 halves.
  // One column per thread: 256 B contiguous = 16 async b128 DMAs.
  // INST_OFFSET applies to both the LDS and global addresses (ISA 15.18.3).
  auto fill = [&](int g) {
    const int cb = g / NKCH;
    const int kc = (g - cb * NKCH) * KCHUNK;
    const _Float16* gp =
        zh + (size_t)(splitBase + cb * COLS_PER_TILE + tid) * DDIM + kc;
    const uint32_t loff = (uint32_t)(uintptr_t)&lb[g & 1][tid * LDS_STRIDE];
#pragma unroll
    for (int q = 0; q < 16; ++q)
      asm volatile("global_load_async_to_lds_b128 %0, %1, off offset:%2"
                   :: "v"(loff), "v"(gp), "i"(q * 16) : "memory");
  };

  // Software-pipelined A fragment (ISA 16-bit A 16x32 layout):
  // lane l: halves 0-7 = K kb..kb+7, halves 8-15 = K kb+16..kb+23 (hi adds 8)
  FragU aCur, aNxt;
  aCur.h[0] = *(const v8h*)(aBase);
  aCur.h[1] = *(const v8h*)(aBase + 16);
  int kbN = KSTEP;

  v8f acc[8];
  fill(0);

  for (int g = 0; g < NCHUNK; ++g) {
    const int cb      = g / NKCH;
    const int kcs     = g - cb * NKCH;
    const int colBase = splitBase + cb * COLS_PER_TILE;

    if (kcs == 0) {
#pragma unroll
      for (int t = 0; t < 8; ++t) acc[t] = (v8f){0.f,0.f,0.f,0.f,0.f,0.f,0.f,0.f};
    }

    if (g + 1 < NCHUNK) {
      fill(g + 1);   // overlap next DMA with this chunk's math
      asm volatile("s_wait_asynccnt 0x10" ::: "memory");  // chunk g landed
    } else {
      asm volatile("s_wait_asynccnt 0x0" ::: "memory");
    }
    __syncthreads();

    const _Float16* lbase = &lb[g & 1][0];
#pragma unroll
    for (int ks = 0; ks < 4; ++ks) {
      // prefetch next A fragment from global (L2-resident)
      aNxt.h[0] = *(const v8h*)(aBase + kbN);
      aNxt.h[1] = *(const v8h*)(aBase + kbN + 16);

      // preload all 8 B fragments (lane n = col, K split by lane half)
      FragU bf[8];
#pragma unroll
      for (int t = 0; t < 8; ++t) {
        const _Float16* bs =
            lbase + (t * 16 + l16) * LDS_STRIDE + ks * KSTEP + hi * 8;
        bf[t].h[0] = *(const v8h*)(bs);
        bf[t].h[1] = *(const v8h*)(bs + 16);
      }

      // 8 back-to-back WMMAs
#pragma unroll
      for (int t = 0; t < 8; ++t)
        acc[t] = __builtin_amdgcn_wmma_f32_16x16x32_f16(
            false, aCur.v, false, bf[t].v, (short)0, acc[t], false, false);

      aCur = aNxt;
      kbN += KSTEP; if (kbN == DDIM) kbN = 0;
    }
    __syncthreads();   // all waves done reading before buffer is re-filled

    if (kcs == NKCH - 1) {
      // Online softmax update over this 16x128 block of sim.
      // C/D layout: VGPR i of lane -> row rowBase + i + 8*hi, col colBase+t*16+l16
#pragma unroll
      for (int i = 0; i < 8; ++i) {
        const int row = rowBase + i + 8 * hi;
        const int pos = (row < N_ROWS) ? row + N_ROWS : row - N_ROWS;
        float vv[8];
        float lm = NEG_BIG;
#pragma unroll
        for (int t = 0; t < 8; ++t) {
          const int col = colBase + t * 16 + l16;
          float v = acc[t][i] * INV_T;
          if (col == row) v = NEG_BIG;              // mask diagonal
          if (col == pos) posA[i] += v;             // capture positive logit
          vv[t] = v;
          lm = fmaxf(lm, v);
        }
        lm = fmaxf(lm, __shfl_xor(lm, 1, 32));
        lm = fmaxf(lm, __shfl_xor(lm, 2, 32));
        lm = fmaxf(lm, __shfl_xor(lm, 4, 32));
        lm = fmaxf(lm, __shfl_xor(lm, 8, 32));
        const float mNew = fmaxf(runM[i], lm);
        float es = 0.f;
#pragma unroll
        for (int t = 0; t < 8; ++t) es += __expf(vv[t] - mNew);
        es += __shfl_xor(es, 1, 32);
        es += __shfl_xor(es, 2, 32);
        es += __shfl_xor(es, 4, 32);
        es += __shfl_xor(es, 8, 32);
        runS[i] = runS[i] * __expf(runM[i] - mNew) + es;
        runM[i] = mNew;
      }
    }
  }

  // Reduce pos across the 16-lane half (at most one contribution per row/split)
#pragma unroll
  for (int i = 0; i < 8; ++i) {
    float p = posA[i];
    p += __shfl_xor(p, 1, 32);
    p += __shfl_xor(p, 2, 32);
    p += __shfl_xor(p, 4, 32);
    p += __shfl_xor(p, 8, 32);
    posA[i] = p;
  }

  if (l16 == 0) {
#pragma unroll
    for (int i = 0; i < 8; ++i) {
      const int row = rowBase + i + 8 * hi;
      const size_t idx = (size_t)blockIdx.y * N2 + row;
      pM[idx]   = runM[i];
      pS[idx]   = runS[i];
      pPos[idx] = posA[i];
    }
  }
}

// ---------------------------------------------------------------------------
// Kernel 3: merge column-split partials, mean(lse - pos) -> scalar
// ---------------------------------------------------------------------------
__global__ __launch_bounds__(256) void ntxent_finish_kernel(
    const float* __restrict__ pM, const float* __restrict__ pS,
    const float* __restrict__ pPos, float* __restrict__ out) {
  __shared__ float red[256];
  float acc = 0.f;
  for (int row = threadIdx.x; row < N2; row += 256) {
    float M = NEG_BIG;
#pragma unroll
    for (int c = 0; c < NSPLIT; ++c) M = fmaxf(M, pM[c * N2 + row]);
    float s = 0.f, pos = 0.f;
#pragma unroll
    for (int c = 0; c < NSPLIT; ++c) {
      s   += pS[c * N2 + row] * __expf(pM[c * N2 + row] - M);
      pos += pPos[c * N2 + row];
    }
    acc += (M + __logf(s)) - pos;
  }
  red[threadIdx.x] = acc;
  __syncthreads();
  for (int s = 128; s > 0; s >>= 1) {
    if (threadIdx.x < s) red[threadIdx.x] += red[threadIdx.x + s];
    __syncthreads();
  }
  if (threadIdx.x == 0) out[0] = red[0] / (float)N2;
}

// ---------------------------------------------------------------------------
extern "C" void kernel_launch(void* const* d_in, const int* in_sizes, int n_in,
                              void* d_out, int out_size, void* d_ws, size_t ws_size,
                              hipStream_t stream) {
  (void)in_sizes; (void)n_in; (void)out_size; (void)ws_size;
  const float* z1 = (const float*)d_in[0];
  const float* z2 = (const float*)d_in[1];
  float* out = (float*)d_out;

  char* ws = (char*)d_ws;
  _Float16* zh = (_Float16*)ws;                       // 8192*768*2 = 12.58 MB
  const size_t zhBytes = (size_t)N2 * DDIM * sizeof(_Float16);
  float* pM   = (float*)(ws + zhBytes);               // NSPLIT * 8192 floats
  float* pS   = pM   + (size_t)NSPLIT * N2;
  float* pPos = pS   + (size_t)NSPLIT * N2;

  ntxent_norm_kernel<<<N2, 256, 0, stream>>>(z1, z2, zh);

  dim3 g2(N2 / ROWS_PER_BLK, NSPLIT);                 // 128 x 4 blocks
  ntxent_sim_kernel<<<g2, 128, 0, stream>>>(zh, pM, pS, pPos);

  ntxent_finish_kernel<<<1, 256, 0, stream>>>(pM, pS, pPos, out);
}
